// LSTM_cell_2190433321306
// MI455X (gfx1250) — matrix operand order repaired and checked by reading, not compiled, not run on hardware
//
#include <hip/hip_runtime.h>
#include <hip/hip_bf16.h>
#include <math.h>

// LSTM on MI455X (gfx1250): bf16 WMMA GEMMs with f32 accumulate.
// xproj = one big parallel GEMM; recurrence = 512 sequential fused GEMM+gate kernels.
// Global->LDS staging uses CDNA5 async loads (ASYNCcnt) with double-buffered LDS.

typedef __bf16 bf16_t;
typedef __attribute__((ext_vector_type(16))) __bf16 v16bf;
typedef __attribute__((ext_vector_type(8)))  float  v8f;

#define T_STEPS 512
#define BATCH   64
#define IN_DIM  1024
#define HID_DIM 1024
#define G4      4096   // 4 * HID
#define LDSP    40     // row pitch (elems) for 32-wide K slab, padded to stagger banks

union FragU { v16bf v; uint4 q[2]; };

// CDNA5 async global->LDS copy, 16 bytes. VDST = LDS byte address (low 32 bits of
// generic LDS pointer per ISA aperture rules), VADDR = 64-bit global address.
__device__ __forceinline__ void async_b128(const bf16_t* gsrc, const bf16_t* lds_dst) {
    unsigned int loff = (unsigned int)(unsigned long long)lds_dst;
    asm volatile("global_load_async_to_lds_b128 %0, %1, off"
                 :: "v"(loff), "v"(gsrc)
                 : "memory");
}
__device__ __forceinline__ void wait_async0() {
    asm volatile("s_wait_asynccnt 0" ::: "memory");
}

// A fragment 16x32 bf16 (ISA 7.12.2): lane<16 -> K {0..7,16..23}; lane>=16 -> K {8..15,24..31}
__device__ __forceinline__ v16bf load_a_frag(const bf16_t* As, int row, int lane) {
    int k0 = (lane & 16) ? 8 : 0;
    FragU u;
    u.q[0] = *(const uint4*)(As + row * LDSP + k0);
    u.q[1] = *(const uint4*)(As + row * LDSP + k0 + 16);
    return u.v;
}
// B fragment 32x16 bf16: lane<16 -> K {0..7,16..23}, lane>=16 -> K {8..15,24..31}, col = lane&15 (Bs stored [n][k])
__device__ __forceinline__ v16bf load_b_frag(const bf16_t* Bs, int n, int lane) {
    int k0 = (lane & 16) ? 8 : 0;
    FragU u;
    u.q[0] = *(const uint4*)(Bs + n * LDSP + k0);
    u.q[1] = *(const uint4*)(Bs + n * LDSP + k0 + 16);
    return u.v;
}

__device__ __forceinline__ float sigm_(float x) { return 1.0f / (1.0f + __expf(-x)); }

// ---------------- precompute kernels ----------------

__global__ void cvt_f32_bf16(const float* __restrict__ in, bf16_t* __restrict__ out, size_t n) {
    for (size_t i = (size_t)blockIdx.x * blockDim.x + threadIdx.x; i < n;
         i += (size_t)gridDim.x * blockDim.x)
        out[i] = (bf16_t)in[i];
}

// Pack Wx = [Wix;Wfx;Wox;Wcx] and Wh = [Wih;Wfh;Woh;Wch] to bf16 [4096][1024]; bias = bx+bh.
__global__ void pack_weights(const float* __restrict__ Wix, const float* __restrict__ Wfx,
                             const float* __restrict__ Wox, const float* __restrict__ Wcx,
                             const float* __restrict__ Wih, const float* __restrict__ Wfh,
                             const float* __restrict__ Woh, const float* __restrict__ Wch,
                             const float* __restrict__ bix, const float* __restrict__ bih,
                             const float* __restrict__ bfx, const float* __restrict__ bfh,
                             const float* __restrict__ box_, const float* __restrict__ boh,
                             const float* __restrict__ bcx, const float* __restrict__ bch,
                             bf16_t* __restrict__ Wxb, bf16_t* __restrict__ Whb,
                             float* __restrict__ bias) {
    size_t idx = (size_t)blockIdx.x * blockDim.x + threadIdx.x;   // < 4096*1024
    int row = (int)(idx >> 10);
    int g = row >> 10;
    int r = row & 1023;
    int k = (int)(idx & 1023);
    const float* wx[4] = {Wix, Wfx, Wox, Wcx};
    const float* wh[4] = {Wih, Wfh, Woh, Wch};
    Wxb[idx] = (bf16_t)wx[g][(size_t)r * IN_DIM + k];
    Whb[idx] = (bf16_t)wh[g][(size_t)r * HID_DIM + k];
    if (idx < G4) {
        const float* bx[4] = {bix, bfx, box_, bcx};
        const float* bh[4] = {bih, bfh, boh, bch};
        int bg = (int)(idx >> 10), bj = (int)(idx & 1023);
        bias[idx] = bx[bg][bj] + bh[bg][bj];
    }
}

__global__ void init_state(const float* __restrict__ H0, const float* __restrict__ C0,
                           bf16_t* __restrict__ hb0, float* __restrict__ c) {
    int i = blockIdx.x * blockDim.x + threadIdx.x;   // 65536 threads
    hb0[i] = (bf16_t)H0[i];
    c[i] = C0[i];
}

// ---------------- xproj GEMM: [32768,1024] @ [1024,4096] + bias ----------------
// 256 threads = 8 waves; 64x64 tile; wave (w&3, w>>2) -> 16x32 sub-tile (2 accumulators).
// Double-buffered LDS, async global->LDS, one barrier per K slab.

__global__ void gemm_xproj(const bf16_t* __restrict__ Xb, const bf16_t* __restrict__ Wxb,
                           const float* __restrict__ bias, float* __restrict__ xproj) {
    __shared__ bf16_t As[2][64 * LDSP];
    __shared__ bf16_t Bs[2][64 * LDSP];
    int tid = threadIdx.x;
    int lane = tid & 31, wave = tid >> 5;
    int mb = (wave & 3) * 16;
    int nh = (wave >> 2) * 32;
    int mTile = blockIdx.x * 64;
    int nTile = blockIdx.y * 64;

    v8f acc0 = {}, acc1 = {};
    int lr = tid >> 2;            // tile row 0..63
    int lk = (tid & 3) * 8;       // k offset 0,8,16,24 (16B granules)
    const bf16_t* gA = Xb + (size_t)(mTile + lr) * IN_DIM + lk;
    const bf16_t* gB = Wxb + (size_t)(nTile + lr) * IN_DIM + lk;
    int lidx = lr * LDSP + lk;

    // prologue: fetch slab 0 into buffer 0
    async_b128(gA, &As[0][lidx]);
    async_b128(gB, &Bs[0][lidx]);

    const int NK = IN_DIM / 32;
    for (int kbi = 0; kbi < NK; ++kbi) {
        int cur = kbi & 1;
        wait_async0();       // my async writes done
        __syncthreads();     // everyone's writes visible; prior reads of other buf done
        if (kbi + 1 < NK) {
            async_b128(gA + (size_t)(kbi + 1) * 32, &As[cur ^ 1][lidx]);
            async_b128(gB + (size_t)(kbi + 1) * 32, &Bs[cur ^ 1][lidx]);
        }
        v16bf a  = load_a_frag(&As[cur][0], mb + (lane & 15), lane);
        v16bf b0 = load_b_frag(&Bs[cur][0], nh + (lane & 15), lane);
        v16bf b1 = load_b_frag(&Bs[cur][0], nh + 16 + (lane & 15), lane);
        acc0 = __builtin_amdgcn_wmma_f32_16x16x32_bf16(false, a, false, b0, (short)0, acc0, false, false);
        acc1 = __builtin_amdgcn_wmma_f32_16x16x32_bf16(false, a, false, b1, (short)0, acc1, false, false);
    }
    int mlo = mb + ((lane & 16) ? 8 : 0);
    int ncol = lane & 15;
    for (int r = 0; r < 8; ++r) {
        size_t gm = (size_t)(mTile + mlo + r);
        int gn0 = nTile + nh + ncol;
        xproj[gm * G4 + gn0]      = acc0[r] + bias[gn0];
        xproj[gm * G4 + gn0 + 16] = acc1[r] + bias[gn0 + 16];
    }
}

// ---------------- one LSTM timestep: fused h@Wh^T (all 4 gates) + gates ----------------
// 16 workgroups, each owns H-columns [nb, nb+64) of ALL 4 gate blocks -> finishes h locally.

__global__ void lstm_step(const bf16_t* __restrict__ hb_in, bf16_t* __restrict__ hb_out,
                          const bf16_t* __restrict__ Whb, const float* __restrict__ xproj_t,
                          float* __restrict__ c, float* __restrict__ out_t) {
    __shared__ bf16_t As[2][64 * LDSP];
    __shared__ bf16_t Bs[2][4][64 * LDSP];
    int tid = threadIdx.x;
    int lane = tid & 31, wave = tid >> 5;
    int mb = (wave & 3) * 16;
    int nh = (wave >> 2) * 32;
    int nb = blockIdx.x * 64;

    v8f acc[4][2];
    v8f z = {};
    for (int g = 0; g < 4; ++g) { acc[g][0] = z; acc[g][1] = z; }

    int lr = tid >> 2;
    int lk = (tid & 3) * 8;
    int lidx = lr * LDSP + lk;
    const bf16_t* gA = hb_in + (size_t)lr * HID_DIM + lk;
    const bf16_t* gB0 = Whb + (size_t)(0 * HID_DIM + nb + lr) * HID_DIM + lk;
    const bf16_t* gB1 = Whb + (size_t)(1 * HID_DIM + nb + lr) * HID_DIM + lk;
    const bf16_t* gB2 = Whb + (size_t)(2 * HID_DIM + nb + lr) * HID_DIM + lk;
    const bf16_t* gB3 = Whb + (size_t)(3 * HID_DIM + nb + lr) * HID_DIM + lk;

    // prologue: slab 0 -> buffer 0
    async_b128(gA,  &As[0][lidx]);
    async_b128(gB0, &Bs[0][0][lidx]);
    async_b128(gB1, &Bs[0][1][lidx]);
    async_b128(gB2, &Bs[0][2][lidx]);
    async_b128(gB3, &Bs[0][3][lidx]);

    const int NK = HID_DIM / 32;
    for (int kbi = 0; kbi < NK; ++kbi) {
        int cur = kbi & 1;
        wait_async0();
        __syncthreads();
        if (kbi + 1 < NK) {
            size_t ko = (size_t)(kbi + 1) * 32;
            async_b128(gA + ko,  &As[cur ^ 1][lidx]);
            async_b128(gB0 + ko, &Bs[cur ^ 1][0][lidx]);
            async_b128(gB1 + ko, &Bs[cur ^ 1][1][lidx]);
            async_b128(gB2 + ko, &Bs[cur ^ 1][2][lidx]);
            async_b128(gB3 + ko, &Bs[cur ^ 1][3][lidx]);
        }
        v16bf a = load_a_frag(&As[cur][0], mb + (lane & 15), lane);   // reused by all 4 gates
        #pragma unroll
        for (int g = 0; g < 4; ++g) {
            v16bf b0 = load_b_frag(&Bs[cur][g][0], nh + (lane & 15), lane);
            v16bf b1 = load_b_frag(&Bs[cur][g][0], nh + 16 + (lane & 15), lane);
            acc[g][0] = __builtin_amdgcn_wmma_f32_16x16x32_bf16(false, a, false, b0, (short)0, acc[g][0], false, false);
            acc[g][1] = __builtin_amdgcn_wmma_f32_16x16x32_bf16(false, a, false, b1, (short)0, acc[g][1], false, false);
        }
    }

    int mlo = mb + ((lane & 16) ? 8 : 0);
    int ncol = lane & 15;
    #pragma unroll
    for (int f = 0; f < 2; ++f) {
        #pragma unroll
        for (int r = 0; r < 8; ++r) {
            int m   = mlo + r;                         // batch row
            int col = nb + nh + f * 16 + ncol;         // column within H
            size_t xrow = (size_t)m * G4;
            float gi = acc[0][f][r] + xproj_t[xrow + 0 * HID_DIM + col];
            float gf = acc[1][f][r] + xproj_t[xrow + 1 * HID_DIM + col];
            float go = acc[2][f][r] + xproj_t[xrow + 2 * HID_DIM + col];
            float gc = acc[3][f][r] + xproj_t[xrow + 3 * HID_DIM + col];
            float iv = sigm_(gi), fv = sigm_(gf), ov = sigm_(go);
            float ct = tanhf(gc);
            size_t idx = (size_t)m * HID_DIM + col;
            float cn = fv * c[idx] + iv * ct;
            float h  = ov * tanhf(cn);
            c[idx] = cn;
            out_t[idx] = h;
            hb_out[idx] = (bf16_t)h;
        }
    }
}

__global__ void finalize(const float* __restrict__ c, float* __restrict__ out) {
    int i = blockIdx.x * blockDim.x + threadIdx.x;   // 65536 threads
    size_t base = (size_t)T_STEPS * BATCH * HID_DIM;
    out[base + i] = out[(size_t)(T_STEPS - 1) * BATCH * HID_DIM + i];   // Hf
    out[base + (size_t)BATCH * HID_DIM + i] = c[i];                     // Cf
}

// ---------------- host ----------------

extern "C" void kernel_launch(void* const* d_in, const int* in_sizes, int n_in,
                              void* d_out, int out_size, void* d_ws, size_t ws_size,
                              hipStream_t stream) {
    const float* inputs = (const float*)d_in[0];
    const float* H0 = (const float*)d_in[1];
    const float* C0 = (const float*)d_in[2];
    const float* Wix = (const float*)d_in[3];  const float* bix = (const float*)d_in[4];
    const float* Wih = (const float*)d_in[5];  const float* bih = (const float*)d_in[6];
    const float* Wfx = (const float*)d_in[7];  const float* bfx = (const float*)d_in[8];
    const float* Wfh = (const float*)d_in[9];  const float* bfh = (const float*)d_in[10];
    const float* Wox = (const float*)d_in[11]; const float* box_ = (const float*)d_in[12];
    const float* Woh = (const float*)d_in[13]; const float* boh = (const float*)d_in[14];
    const float* Wcx = (const float*)d_in[15]; const float* bcx = (const float*)d_in[16];
    const float* Wch = (const float*)d_in[17]; const float* bch = (const float*)d_in[18];
    float* out = (float*)d_out;

    // workspace carve-up
    char* p = (char*)d_ws;
    bf16_t* xb  = (bf16_t*)p; p += (size_t)T_STEPS * BATCH * IN_DIM * sizeof(bf16_t);
    bf16_t* wxb = (bf16_t*)p; p += (size_t)G4 * IN_DIM * sizeof(bf16_t);
    bf16_t* whb = (bf16_t*)p; p += (size_t)G4 * HID_DIM * sizeof(bf16_t);
    float*  bias = (float*)p; p += (size_t)G4 * sizeof(float);
    bf16_t* hb  = (bf16_t*)p; p += (size_t)2 * BATCH * HID_DIM * sizeof(bf16_t);
    float*  cst = (float*)p;  p += (size_t)BATCH * HID_DIM * sizeof(float);
    float*  xproj = (float*)p;   // [T][B][4H] f32

    size_t nin = (size_t)T_STEPS * BATCH * IN_DIM;
    cvt_f32_bf16<<<4096, 256, 0, stream>>>(inputs, xb, nin);
    pack_weights<<<(G4 * IN_DIM) / 256, 256, 0, stream>>>(
        Wix, Wfx, Wox, Wcx, Wih, Wfh, Woh, Wch,
        bix, bih, bfx, bfh, box_, boh, bcx, bch, wxb, whb, bias);
    init_state<<<(BATCH * HID_DIM) / 256, 256, 0, stream>>>(H0, C0, hb, cst);

    gemm_xproj<<<dim3((T_STEPS * BATCH) / 64, G4 / 64), 256, 0, stream>>>(xb, wxb, bias, xproj);

    for (int t = 0; t < T_STEPS; ++t) {
        const bf16_t* hin = hb + (size_t)(t & 1) * BATCH * HID_DIM;
        bf16_t* hout      = hb + (size_t)((t + 1) & 1) * BATCH * HID_DIM;
        lstm_step<<<HID_DIM / 64, 256, 0, stream>>>(
            hin, hout, whb, xproj + (size_t)t * BATCH * G4,
            cst, out + (size_t)t * BATCH * HID_DIM);
    }
    finalize<<<(BATCH * HID_DIM) / 256, 256, 0, stream>>>(cst, out);
}
